// painn_85341000171730
// MI455X (gfx1250) — matrix operand order repaired and checked by reading, not compiled, not run on hardware
//
#include <hip/hip_runtime.h>

// ---------------------------------------------------------------------------
// PaiNN-style GNN for MI455X (gfx1250, wave32, WMMA).
// All GEMMs run through one f16-WMMA tiled kernel (v_wmma_f32_16x16x32_f16),
// 64x128 block tile, async global->LDS staging for the weight tile.
// ---------------------------------------------------------------------------

typedef __attribute__((ext_vector_type(16))) _Float16 v16h;
typedef __attribute__((ext_vector_type(8)))  _Float16 h8;
typedef __attribute__((ext_vector_type(4)))  _Float16 h4;
typedef __attribute__((ext_vector_type(8)))  float    v8f;
typedef __attribute__((ext_vector_type(4)))  float    f4;

#define TILE_M 64
#define TILE_N 128
#define TILE_K 32

// gfx1250 async global->LDS copy (ASYNCcnt-tracked). LDS address = low 32 bits
// of the generic pointer (ISA: flat LDS aperture maps addr[31:0] -> LDS).
#define USE_ASYNC_LDS 1

// ---------------------------------------------------------------------------
// C[M,Nout] = epilogue( (GATHER? A[gidx[m]] : A[m]) @ Wt^T + bias )
// A: f32 row-major (lda==K). Wt: f16 *transposed* weights, Nout x K row-major.
// ACT: SiLU. SCALE: per-column scale (gate = w * phi).
// Block: 128 threads = 4 waves; wave w owns rows [w*16, w*16+16) of the 64-row
// tile and all 128 columns as 8 WMMA accumulator tiles.
// ---------------------------------------------------------------------------
template<int GATHER, int ACT, int SCALE>
__global__ __launch_bounds__(128)
void k_wmma_gemm(const float* __restrict__ A,
                 const _Float16* __restrict__ Wt,
                 const float* __restrict__ bias,
                 const float* __restrict__ colscale,
                 float* __restrict__ C,
                 const int* __restrict__ gidx,
                 int Mrows, int K, int Nout)
{
    __shared__ __attribute__((aligned(16))) _Float16 sA[TILE_M * TILE_K]; // [row][k]
    __shared__ __attribute__((aligned(16))) _Float16 sB[TILE_N * TILE_K]; // [n][k]

    const int tid  = threadIdx.x;
    const int wave = tid >> 5;
    const int lane = tid & 31;
    const int grp  = lane >> 4;   // lane half: 0 or 1
    const int l16  = lane & 15;

    const int rBase = blockIdx.x * TILE_M;
    const int cBase = blockIdx.y * TILE_N;
    const int wRow  = rBase + wave * 16;

    // Accumulators init with bias (bias is per-column, identical across rows).
    v8f acc[8];
#pragma unroll
    for (int ct = 0; ct < 8; ++ct) {
        float bv = bias ? bias[cBase + ct * 16 + l16] : 0.0f;
#pragma unroll
        for (int r = 0; r < 8; ++r) acc[ct][r] = bv;
    }

    // Pre-resolve the 4 source rows this thread stages (branchless clamp).
    int srcRow[4];
#pragma unroll
    for (int i = 0; i < 4; ++i) {
        int g  = tid + i * 128;          // 0..511 float4 groups
        int r  = g >> 3;                 // tile row 0..63
        int gr = rBase + r;
        int rc = gr < Mrows ? gr : (Mrows - 1);
        srcRow[i] = GATHER ? gidx[rc] : rc;
    }

    const int ksteps = K / TILE_K;
    for (int ks = 0; ks < ksteps; ++ks) {
        const int k0 = ks * TILE_K;

        // ---- stage B tile: Wt rows are contiguous in k -> pure 16B copies ----
#pragma unroll
        for (int i = 0; i < 4; ++i) {
            int g  = tid + i * 128;      // 0..511 h8 groups
            int n  = g >> 2;             // 0..127
            int ko = (g & 3) * 8;        // 0,8,16,24
            const _Float16* gp = &Wt[(size_t)(cBase + n) * K + (k0 + ko)];
            _Float16* lp = &sB[n * TILE_K + ko];
#if USE_ASYNC_LDS
            unsigned      lds_off = (unsigned)(uintptr_t)lp;
            unsigned long long ga = (unsigned long long)(uintptr_t)gp;
            asm volatile("global_load_async_to_lds_b128 %0, %1, off"
                         :: "v"(lds_off), "v"(ga) : "memory");
#else
            *(h8*)lp = *(const h8*)gp;
#endif
        }

        // ---- stage A tile: f32 -> f16 through VGPRs (needs conversion) ----
#pragma unroll
        for (int i = 0; i < 4; ++i) {
            int g  = tid + i * 128;
            int r  = g >> 3;
            int kq = (g & 7) * 4;        // 0..28
            f4 a = *(const f4*)&A[(size_t)srcRow[i] * K + (k0 + kq)];
            h4 hv;
            hv[0] = (_Float16)a[0]; hv[1] = (_Float16)a[1];
            hv[2] = (_Float16)a[2]; hv[3] = (_Float16)a[3];
            *(h4*)&sA[r * TILE_K + kq] = hv;
        }

#if USE_ASYNC_LDS
        asm volatile("s_wait_asynccnt 0x0" ::: "memory");
#endif
        __syncthreads();

        // A fragment (16x32 f16): lane = M%16; K interleaved by lane half.
        union Frag { v16h h; unsigned u[8]; };
        Frag fa;
        {
            const unsigned* pa = (const unsigned*)sA;
            const int arow = wave * 16 + l16;
#pragma unroll
            for (int vv = 0; vv < 8; ++vv) {
                int k = ((vv & 3) * 2) + ((vv >> 2) * 16) + grp * 8;
                fa.u[vv] = pa[(arow * TILE_K + k) >> 1];
            }
        }
        // Preload ALL 8 B fragments, then issue the WMMA chain back-to-back
        // (lets the compiler issue the ds_loads ahead and use staggered waits).
        const unsigned* pb = (const unsigned*)sB;
        Frag fb[8];
#pragma unroll
        for (int ct = 0; ct < 8; ++ct) {
            const int nrow = ct * 16 + l16;
#pragma unroll
            for (int vv = 0; vv < 8; ++vv) {
                int k = ((vv & 3) * 2) + ((vv >> 2) * 16) + grp * 8;
                fb[ct].u[vv] = pb[(nrow * TILE_K + k) >> 1];
            }
        }
#pragma unroll
        for (int ct = 0; ct < 8; ++ct) {
            acc[ct] = __builtin_amdgcn_wmma_f32_16x16x32_f16(
                false, fa.h, false, fb[ct].h, (short)0, acc[ct], false, false);
        }
        __syncthreads();
    }

    // Epilogue (C/D layout: VGPR r holds rows r / r+8 by lane half).
#pragma unroll
    for (int ct = 0; ct < 8; ++ct) {
        int   col = cBase + ct * 16 + l16;
        float cs  = SCALE ? colscale[col] : 1.0f;
#pragma unroll
        for (int r = 0; r < 8; ++r) {
            int row = wRow + r + grp * 8;
            if (row < Mrows) {
                float x = acc[ct][r];
                if (ACT) x = x / (1.0f + __expf(-x));   // SiLU
                C[(size_t)row * Nout + col] = x * cs;
            }
        }
    }
}

// ---------------------------------------------------------------------------
// Small helper kernels
// ---------------------------------------------------------------------------

// Cast + transpose weights: W (K x Nout, f32) -> Wt (Nout x K, f16)
__global__ void k_cast_transpose(const float* __restrict__ W, _Float16* __restrict__ Wt,
                                 int K, int Nout)
{
    int idx = blockIdx.x * blockDim.x + threadIdx.x;
    if (idx < K * Nout) {
        int k = idx / Nout, n = idx % Nout;
        Wt[(size_t)n * K + k] = (_Float16)W[idx];
    }
}

__global__ void k_init_node(const int* __restrict__ z, const float* __restrict__ emb,
                            float* __restrict__ s, float* __restrict__ v)
{
    int i = blockIdx.x, d = threadIdx.x;
    s[(size_t)i * 128 + d] = emb[(size_t)z[i] * 128 + d];
    for (int c = 0; c < 3; ++c) v[((size_t)i * 3 + c) * 128 + d] = 0.0f;
}

// Reference quirk reproduced: rbf/fcut/w/r_hat come from edge 0 only.
__global__ void k_edge_w(const float* __restrict__ pos,
                         const int* __restrict__ idx_i, const int* __restrict__ idx_j,
                         const float* __restrict__ ww, const float* __restrict__ wb,
                         float* __restrict__ wvec /*384*/, float* __restrict__ rhat /*3*/)
{
    __shared__ float fcut[20];
    const int i0 = idx_i[0], j0 = idx_j[0];
    const float rx = pos[i0 * 3 + 0] - pos[j0 * 3 + 0];
    const float ry = pos[i0 * 3 + 1] - pos[j0 * 3 + 1];
    const float rz = pos[i0 * 3 + 2] - pos[j0 * 3 + 2];
    const float rn = sqrtf(rx * rx + ry * ry + rz * rz);
    const float PI = 3.14159265358979323846f;
    const int t = threadIdx.x;
    if (t < 20) {
        float rbf = sinf((float)(t + 1) * PI * 0.5f * rn) / rn;
        fcut[t] = (rbf <= 2.0f) ? 0.5f * (cosf(PI * rbf * 0.5f) + 1.0f) : 0.0f;
    }
    __syncthreads();
    for (int o = t; o < 384; o += blockDim.x) {
        float a = wb[o];
        for (int k = 0; k < 20; ++k) a += fcut[k] * ww[k * 384 + o];
        wvec[o] = a;
    }
    if (t == 0) { rhat[0] = rx / rn; rhat[1] = ry / rn; rhat[2] = rz / rn; }
}

// Segment-sum scatter of edge messages (reads v snapshot, atomically updates s, v).
__global__ void k_edge_scatter(const float* __restrict__ gate,
                               const float* __restrict__ v_old,
                               const int* __restrict__ idx_i, const int* __restrict__ idx_j,
                               const float* __restrict__ rhat,
                               float* __restrict__ s, float* __restrict__ v, int E)
{
    int e = blockIdx.x;
    if (e >= E) return;
    int d = threadIdx.x;
    int i = idx_i[e], j = idx_j[e];
    float g0 = gate[(size_t)e * 384 + d];
    float g1 = gate[(size_t)e * 384 + 128 + d];
    float g2 = gate[(size_t)e * 384 + 256 + d];
    atomicAdd(&s[(size_t)i * 128 + d], g1);
    for (int c = 0; c < 3; ++c) {
        float ve = g0 * v_old[((size_t)j * 3 + c) * 128 + d] + g2 * rhat[c];
        atomicAdd(&v[((size_t)i * 3 + c) * 128 + d], ve);
    }
}

__global__ void k_rownorm_concat(const float* __restrict__ Vv, const float* __restrict__ s,
                                 float* __restrict__ a_in)
{
    int i = blockIdx.x, d = threadIdx.x;
    float x0 = Vv[((size_t)i * 3 + 0) * 128 + d];
    float x1 = Vv[((size_t)i * 3 + 1) * 128 + d];
    float x2 = Vv[((size_t)i * 3 + 2) * 128 + d];
    a_in[(size_t)i * 256 + d]       = sqrtf(x0 * x0 + x1 * x1 + x2 * x2);
    a_in[(size_t)i * 256 + 128 + d] = s[(size_t)i * 128 + d];
}

__global__ void k_node_update(const float* __restrict__ u, const float* __restrict__ Vv,
                              const float* __restrict__ a,
                              float* __restrict__ s, float* __restrict__ v)
{
    int i = blockIdx.x, d = threadIdx.x;
    float a0 = a[(size_t)i * 384 + d];
    float a1 = a[(size_t)i * 384 + 128 + d];
    float a2 = a[(size_t)i * 384 + 256 + d];
    float dot = 0.0f;
    for (int c = 0; c < 3; ++c) {
        size_t ix = ((size_t)i * 3 + c) * 128 + d;
        dot += u[ix] * Vv[ix];
        v[ix] += u[ix] * a0;
    }
    s[(size_t)i * 128 + d] += dot * a1 + a2;
}

// out[m] = sum over molecule m's 16 atoms of rowsum(out2[atom,:])
__global__ void k_mol_reduce(const float* __restrict__ o2, float* __restrict__ out)
{
    __shared__ float red[128];
    int m = blockIdx.x, t = threadIdx.x;
    float p = 0.0f;
    for (int a = 0; a < 16; ++a) p += o2[((size_t)(m * 16 + a)) * 128 + t];
    red[t] = p;
    __syncthreads();
    for (int st = 64; st > 0; st >>= 1) {
        if (t < st) red[t] += red[t + st];
        __syncthreads();
    }
    if (t == 0) out[m] = red[0];
}

// ---------------------------------------------------------------------------
// Host driver
// ---------------------------------------------------------------------------
static void launch_gemm(hipStream_t st, const float* A, const int* gather, int Mr, int K,
                        const _Float16* Wt, const float* bias, const float* cs,
                        float* C, int Nout, int act)
{
    dim3 grid((unsigned)((Mr + TILE_M - 1) / TILE_M), (unsigned)(Nout / TILE_N));
    dim3 blk(128);
    if (gather)       k_wmma_gemm<1, 1, 0><<<grid, blk, 0, st>>>(A, Wt, bias, cs, C, gather, Mr, K, Nout);
    else if (cs)      k_wmma_gemm<0, 0, 1><<<grid, blk, 0, st>>>(A, Wt, bias, cs, C, gather, Mr, K, Nout);
    else if (act)     k_wmma_gemm<0, 1, 0><<<grid, blk, 0, st>>>(A, Wt, bias, cs, C, gather, Mr, K, Nout);
    else              k_wmma_gemm<0, 0, 0><<<grid, blk, 0, st>>>(A, Wt, bias, cs, C, gather, Mr, K, Nout);
}

extern "C" void kernel_launch(void* const* d_in, const int* in_sizes, int n_in,
                              void* d_out, int out_size, void* d_ws, size_t ws_size,
                              hipStream_t stream)
{
    const int N = 8192, Mmol = 512, D = 128;
    const int*   z     = (const int*)  d_in[0];
    const float* pos   = (const float*)d_in[1];
    const int*   idx_i = (const int*)  d_in[3];
    const int*   idx_j = (const int*)  d_in[4];
    const float* emb   = (const float*)d_in[5];
    const float* pw1 = (const float*)d_in[6];  const float* pb1 = (const float*)d_in[7];
    const float* pw2 = (const float*)d_in[8];  const float* pb2 = (const float*)d_in[9];
    const float* ww  = (const float*)d_in[10]; const float* wb  = (const float*)d_in[11];
    const float* aw1 = (const float*)d_in[12]; const float* ab1 = (const float*)d_in[13];
    const float* aw2 = (const float*)d_in[14]; const float* ab2 = (const float*)d_in[15];
    const float* vw  = (const float*)d_in[16]; const float* vb  = (const float*)d_in[17];
    const float* uw  = (const float*)d_in[18]; const float* ub  = (const float*)d_in[19];
    const float* ow1 = (const float*)d_in[20]; const float* ob1 = (const float*)d_in[21];
    const float* ow2 = (const float*)d_in[22]; const float* ob2 = (const float*)d_in[23];
    const int E = in_sizes[3];

    // Workspace carve-out
    char* base = (char*)d_ws;
    size_t off = 0;
    auto alloc = [&](size_t bytes) -> char* {
        off = (off + 255) & ~(size_t)255;
        char* p = base + off;
        off += bytes;
        return p;
    };
    _Float16* pw1h = (_Float16*)alloc(sizeof(_Float16) * 128 * 128);
    _Float16* pw2h = (_Float16*)alloc(sizeof(_Float16) * 128 * 384);
    _Float16* aw1h = (_Float16*)alloc(sizeof(_Float16) * 256 * 128);
    _Float16* aw2h = (_Float16*)alloc(sizeof(_Float16) * 128 * 384);
    _Float16* vwh  = (_Float16*)alloc(sizeof(_Float16) * 128 * 128);
    _Float16* uwh  = (_Float16*)alloc(sizeof(_Float16) * 128 * 128);
    _Float16* ow1h = (_Float16*)alloc(sizeof(_Float16) * 128 * 128);
    _Float16* ow2h = (_Float16*)alloc(sizeof(_Float16) * 128 * 128);
    float* s     = (float*)alloc(sizeof(float) * (size_t)N * D);
    float* v     = (float*)alloc(sizeof(float) * (size_t)N * 3 * D);
    float* v_old = (float*)alloc(sizeof(float) * (size_t)N * 3 * D);
    float* wvec  = (float*)alloc(sizeof(float) * 384);
    float* rhat  = (float*)alloc(sizeof(float) * 4);
    float* H     = (float*)alloc(sizeof(float) * (size_t)E * D);
    float* gate  = (float*)alloc(sizeof(float) * (size_t)E * 3 * D);
    float* Vvb   = (float*)alloc(sizeof(float) * (size_t)N * 3 * D);
    float* ubuf  = (float*)alloc(sizeof(float) * (size_t)N * 3 * D);
    float* a_in  = (float*)alloc(sizeof(float) * (size_t)N * 2 * D);
    float* t1    = (float*)alloc(sizeof(float) * (size_t)N * D);
    float* abuf  = (float*)alloc(sizeof(float) * (size_t)N * 3 * D);
    float* t2    = (float*)alloc(sizeof(float) * (size_t)N * D);
    float* o2    = (float*)alloc(sizeof(float) * (size_t)N * D);

    auto cast = [&](const float* src, _Float16* dst, int K, int Nout) {
        int n = K * Nout;
        k_cast_transpose<<<(n + 255) / 256, 256, 0, stream>>>(src, dst, K, Nout);
    };

    // Weights -> f16, transposed (tiny; stay resident in L2)
    cast(pw1, pw1h, 128, 128);  cast(pw2, pw2h, 128, 384);
    cast(aw1, aw1h, 256, 128);  cast(aw2, aw2h, 128, 384);
    cast(vw,  vwh,  128, 128);  cast(uw,  uwh,  128, 128);
    cast(ow1, ow1h, 128, 128);  cast(ow2, ow2h, 128, 128);

    // State init + edge scalars
    k_init_node<<<N, D, 0, stream>>>(z, emb, s, v);
    k_edge_w<<<1, 128, 0, stream>>>(pos, idx_i, idx_j, ww, wb, wvec, rhat);

    for (int it = 0; it < 3; ++it) {
        if (E > 0) {
            // H = silu(s[idx_j] @ pw1 + pb1)
            launch_gemm(stream, s, idx_j, E, D, pw1h, pb1, nullptr, H, D, 1);
            // gate = w * (H @ pw2 + pb2)
            launch_gemm(stream, H, nullptr, E, D, pw2h, pb2, wvec, gate, 3 * D, 0);
            // snapshot v, then scatter ds/dv into s,v
            hipMemcpyAsync(v_old, v, sizeof(float) * (size_t)N * 3 * D,
                           hipMemcpyDeviceToDevice, stream);
            k_edge_scatter<<<E, D, 0, stream>>>(gate, v_old, idx_i, idx_j, rhat, s, v, E);
        }
        // Vv = v @ vw + vb ; u = Vv @ uw + ub   (rows = N*3 spatial channels)
        launch_gemm(stream, v,   nullptr, N * 3, D, vwh, vb, nullptr, Vvb,  D, 0);
        launch_gemm(stream, Vvb, nullptr, N * 3, D, uwh, ub, nullptr, ubuf, D, 0);
        // a_in = [ ||Vv||_c , s ]
        k_rownorm_concat<<<N, D, 0, stream>>>(Vvb, s, a_in);
        // a = silu(a_in @ aw1 + ab1) @ aw2 + ab2
        launch_gemm(stream, a_in, nullptr, N, 2 * D, aw1h, ab1, nullptr, t1, D, 1);
        launch_gemm(stream, t1,   nullptr, N, D, aw2h, ab2, nullptr, abuf, 3 * D, 0);
        // s += <u,Vv>*a1 + a2 ; v += u*a0
        k_node_update<<<N, D, 0, stream>>>(ubuf, Vvb, abuf, s, v);
    }

    // out = silu(s @ ow1 + ob1) @ ow2 + ob2 ; per-molecule sum
    launch_gemm(stream, s,  nullptr, N, D, ow1h, ob1, nullptr, t2, D, 1);
    launch_gemm(stream, t2, nullptr, N, D, ow2h, ob2, nullptr, o2, D, 0);
    k_mol_reduce<<<Mmol, 128, 0, stream>>>(o2, (float*)d_out);
}
